// GraphGeo_76098230550816
// MI455X (gfx1250) — compile-verified, matrix operand order checked by reading, not run on hardware
//
#include <hip/hip_runtime.h>
#include <hip/hip_bf16.h>
#include <math.h>

// ---------------- problem constants ----------------
#define N_LM   3072
#define N_TG   1024
#define NTOT   4096
#define DIN    64
#define DAS    128
#define DZ     64
#define DINNER 256
#define DOUT   2
#define NHEAD  4
#define DKH    128
#define QKDIM  512           // NHEAD*DKH
#define KSEL   1844          // ceil(3072*0.6)
#define NPAD   128           // padded width of [X|Y] operand (66 -> 128)
#define CNF_STEPS 8
static const long long NN = (long long)NTOT * NTOT;

// ---------------- CDNA5 feature detection ----------------
#if defined(__has_builtin)
#  if __has_builtin(__builtin_amdgcn_tensor_load_to_lds)
#    define HAVE_TDM 1
#  endif
#endif
#ifndef HAVE_TDM
#  define HAVE_TDM 0
#endif
#if __has_include(<hip/amd_detail/amd_gfx1250_TDM.h>)
#  define TDM_SIX_ARG 1      // amdgpu-toolchain (therock) 6-arg builtin
#else
#  define TDM_SIX_ARG 0      // ROCm 7.2 5-arg builtin
#endif

// ---------------- WMMA / vector types (gfx1250, wave32) ----------------
typedef __attribute__((ext_vector_type(16))) __bf16 v16bf;
typedef __attribute__((ext_vector_type(8)))  float  v8f;
typedef __attribute__((ext_vector_type(4)))  unsigned int u32x4;
typedef __attribute__((ext_vector_type(4)))  int          i32x4;
typedef __attribute__((ext_vector_type(8)))  int          i32x8;

union FragBF {
    v16bf v;
    unsigned int d[8];
};

__device__ __forceinline__ unsigned short f2bf(float f) {
    unsigned int x = __float_as_uint(f);
    unsigned int r = (x + 0x7fffu + ((x >> 16) & 1u)) >> 16; // RNE
    return (unsigned short)r;
}

__device__ __forceinline__ float softplusf(float x) {
    return x > 20.f ? x : log1pf(expf(x));
}

// ---------------------------------------------------------------------
// TDM: DMA a 64-row x 64-byte tile (row stride in 8-byte units) into LDS.
// D# group0: count=1, lds_addr, 57-bit global addr, type=2 ("image").
// D# group1: data_size=3 (8B), tensor 8x64 == tile 8x64, dim0 stride.
// ---------------------------------------------------------------------
__device__ __forceinline__ void tdm_stage_64x64B(void* lds_dst, const void* gsrc,
                                                 unsigned row_stride_8B)
{
#if HAVE_TDM
    unsigned long long ga = (unsigned long long)(uintptr_t)gsrc;
    unsigned lds_off = (unsigned)(uintptr_t)lds_dst;   // addr[31:0] selects LDS
    u32x4 g0;
    g0[0] = 1u;                                        // count=1, user mode
    g0[1] = lds_off;                                   // lds_addr
    g0[2] = (unsigned)ga;                              // global_addr[31:0]
    g0[3] = (unsigned)((ga >> 32) & 0x01ffffffu) | (2u << 30); // ga[56:32] | type=2
    i32x8 g1;
    g1[0] = 0x00030000;          // wg_mask=0 | data_size=3 (8B) | no flags
    g1[1] = 8 << 16;             // tensor_dim0 = 8 (x8B = 64B row)
    g1[2] = 64 << 16;            // tensor_dim0 hi | tensor_dim1 = 64 rows
    g1[3] = 8 << 16;             // tensor_dim1 hi | tile_dim0 = 8
    g1[4] = 64;                  // tile_dim1 = 64 | tile_dim2 = 0
    g1[5] = (int)row_stride_8B;  // tensor_dim0_stride lo32 (8B units)
    g1[6] = 0;                   // stride hi | tensor_dim1_stride lo
    g1[7] = 0;
    i32x4 z4 = {0, 0, 0, 0};
#  if TDM_SIX_ARG
    i32x8 z8 = {};
    __builtin_amdgcn_tensor_load_to_lds(g0, g1, z4, z4, z8, 0);
#  else
    __builtin_amdgcn_tensor_load_to_lds(g0, g1, z4, z4, 0);
#  endif
#else
    (void)lds_dst; (void)gsrc; (void)row_stride_8B;
#endif
}

// =====================================================================
// WMMA GEMM:  C(M x Ncols, f32, ldc) = alpha * A(M x K, bf16, lda) @ BT(Ncols x K, bf16, ldb)^T
// Block = 128 threads (4 waves) -> 64x64 C tile; wave w owns rows 16w..16w+15
// with 4 accumulators (A fragment reused 4x). Per 32-wide K step the 64x32 A
// and B tiles are staged to LDS by the Tensor Data Mover (double buffered,
// overlapped with WMMA), fragments then read as ds_load_b128 pairs.
// M, Ncols multiples of 64; K multiple of 32.
// =====================================================================
__global__ __launch_bounds__(128)
void wmma_gemm_bf16_tdm(const unsigned short* __restrict__ A, int lda,
                        const unsigned short* __restrict__ BT, int ldb,
                        float* __restrict__ C, int ldc,
                        int K, float alpha)
{
    __shared__ alignas(16) unsigned char lA[2][64 * 64];
    __shared__ alignas(16) unsigned char lB[2][64 * 64];

    const int tid  = threadIdx.x;
    const int wave = tid >> 5;
    const int lane = tid & 31;
    const int r    = lane & 15;
    const int hi   = lane >> 4;          // 0 or 1
    const int offc = (hi ? 8 : 0) * 2;   // byte offset of this half-wave's K chunk
    const int m0   = blockIdx.y * 64;
    const int n0   = blockIdx.x * 64;

    v8f acc[4] = {};

    // stage tiles for K offset k0e into buffer b
    auto stage_pair = [&](int b, int k0e) {
#if HAVE_TDM
        if (wave == 0) {
            tdm_stage_64x64B(lA[b], A  + (size_t)m0 * lda + k0e, (unsigned)(lda >> 2));
            tdm_stage_64x64B(lB[b], BT + (size_t)n0 * ldb + k0e, (unsigned)(ldb >> 2));
        }
#else
        for (int c = tid; c < 512; c += 128) {            // 2 tiles * 256 16B chunks
            int t = c >> 8, cc = c & 255;
            int row = cc >> 2, q = cc & 3;
            const unsigned short* g = t ? (BT + (size_t)(n0 + row) * ldb + k0e)
                                        : (A  + (size_t)(m0 + row) * lda + k0e);
            u32x4* dst = (u32x4*)((t ? lB[b] : lA[b]) + row * 64);
            dst[q] = *((const u32x4*)g + q);
        }
#endif
    };

    stage_pair(0, 0);
#if HAVE_TDM
    if (wave == 0) __builtin_amdgcn_s_wait_tensorcnt(0);
#endif
    __syncthreads();

    int buf = 0;
    for (int k0 = 0; k0 < K; k0 += 32) {
        if (k0 + 32 < K) stage_pair(buf ^ 1, k0 + 32);   // overlap DMA with WMMA

        FragBF a;
        const unsigned char* arow = lA[buf] + (16 * wave + r) * 64;
        *(u32x4*)&a.d[0] = *(const u32x4*)(arow + offc);
        *(u32x4*)&a.d[4] = *(const u32x4*)(arow + 32 + offc);
#pragma unroll
        for (int t = 0; t < 4; ++t) {
            FragBF b;
            const unsigned char* brow = lB[buf] + (16 * t + r) * 64;
            *(u32x4*)&b.d[0] = *(const u32x4*)(brow + offc);
            *(u32x4*)&b.d[4] = *(const u32x4*)(brow + 32 + offc);
            acc[t] = __builtin_amdgcn_wmma_f32_16x16x32_bf16(
                         false, a.v, false, b.v, (short)0, acc[t], false, false);
        }

#if HAVE_TDM
        if (wave == 0) __builtin_amdgcn_s_wait_tensorcnt(0);
#endif
        __syncthreads();
        buf ^= 1;
    }

    const int col  = n0 + r;
    const int rowb = m0 + 16 * wave + (hi ? 8 : 0);
#pragma unroll
    for (int t = 0; t < 4; ++t)
#pragma unroll
        for (int v = 0; v < 8; ++v)
            C[(size_t)(rowb + v) * ldc + col + 16 * t] = acc[t][v] * alpha;
}

// =====================================================================
// small dense GEMM: C(M x Nc, ldc) = act( A(M x K, lda) @ W(K x Nc) + bias )
// act: 0 none, 1 relu, 2 tanh
// =====================================================================
__global__ void sgemm_bias_act(const float* __restrict__ A, int lda,
                               const float* __restrict__ W,
                               const float* __restrict__ bias,
                               float* __restrict__ C, int ldc,
                               int M, int Nc, int K, int act)
{
    const int n = blockIdx.x * blockDim.x + threadIdx.x;
    const int m = blockIdx.y * blockDim.y + threadIdx.y;
    if (n >= Nc || m >= M) return;
    float s = bias ? bias[n] : 0.f;
    const float* a = A + (size_t)m * lda;
    for (int k = 0; k < K; ++k) s += a[k] * W[(size_t)k * Nc + n];
    if (act == 1) s = s > 0.f ? s : 0.f;
    else if (act == 2) s = tanhf(s);
    C[(size_t)m * ldc + n] = s;
}

// ---------------- elementwise / setup kernels ----------------
__global__ void k_zero(float* p, long long n) {
    long long i = blockIdx.x * (long long)blockDim.x + threadIdx.x;
    if (i < n) p[i] = 0.f;
}

__global__ void k_concat_rows(const float* __restrict__ a, int na,
                              const float* __restrict__ b, int nb,
                              int cols, float* __restrict__ out, int zero_b)
{
    long long idx = blockIdx.x * (long long)blockDim.x + threadIdx.x;
    long long tot = (long long)(na + nb) * cols;
    if (idx >= tot) return;
    int r = (int)(idx / cols), c = (int)(idx % cols);
    out[idx] = (r < na) ? a[(size_t)r * cols + c]
                        : (zero_b ? 0.f : b[(size_t)(r - na) * cols + c]);
}

__global__ void k_node_scalars(const float* __restrict__ X,
                               const float* __restrict__ dW, const float* __restrict__ db,
                               const float* __restrict__ kW, const float* __restrict__ kb,
                               const float* __restrict__ dly,
                               float* __restrict__ dist, float* __restrict__ kap)
{
    int i = blockIdx.x * blockDim.x + threadIdx.x;
    if (i >= NTOT) return;
    float a = db[0], b = db[1], g = db[2], kv = kb[0];
    const float* x = X + (size_t)i * DIN;
    for (int d = 0; d < DIN; ++d) {
        float xv = x[d];
        a += xv * dW[d * 3 + 0];
        b += xv * dW[d * 3 + 1];
        g += xv * dW[d * 3 + 2];
        kv += xv * kW[d];
    }
    dist[i] = a * dly[i] + b + g;
    kap[i] = kv > 0.f ? kv : 0.f;
}

__global__ void k_edge_nt(const float* __restrict__ dist, const float* __restrict__ kap,
                          float* __restrict__ adj)
{
    long long idx = blockIdx.x * (long long)blockDim.x + threadIdx.x;
    if (idx >= NN) return;
    int i = (int)(idx >> 12), j = (int)(idx & 4095);
    adj[idx] = expf(-fabsf(dist[j] - dist[i])) * kap[i];
}

__global__ void k_f32_to_bf16(const float* __restrict__ in, unsigned short* __restrict__ out,
                              long long n, float scale)
{
    long long i = blockIdx.x * (long long)blockDim.x + threadIdx.x;
    if (i < n) out[i] = f2bf(in[i] * scale);
}

__global__ void k_transpose_bf16(const float* __restrict__ in, unsigned short* __restrict__ out,
                                 int rows, int cols)
{   // in rows x cols f32  ->  out cols x rows bf16
    long long idx = blockIdx.x * (long long)blockDim.x + threadIdx.x;
    long long tot = (long long)rows * cols;
    if (idx >= tot) return;
    int r = (int)(idx / cols), c = (int)(idx % cols);
    out[(size_t)c * rows + r] = f2bf(in[idx]);
}

__global__ void k_build_inpT(const float* __restrict__ Xc, const float* __restrict__ Yc,
                             unsigned short* __restrict__ out) // out: NPAD x NTOT bf16 (pad 66->NPAD)
{
    long long idx = blockIdx.x * (long long)blockDim.x + threadIdx.x;
    long long tot = (long long)NPAD * NTOT;
    if (idx >= tot) return;
    int c = (int)(idx / NTOT), r = (int)(idx % NTOT);
    float v = 0.f;
    if (c < DIN)          v = Xc[(size_t)r * DIN + c];
    else if (c < DIN + 2) v = Yc[(size_t)r * 2 + (c - DIN)];
    out[idx] = f2bf(v);
}

// ---------------- softmax over score rows, accumulate into adj ----------------
__global__ __launch_bounds__(256)
void k_softmax_acc(const float* __restrict__ S, float* __restrict__ adj, float w)
{
    __shared__ float red[256];
    const int i = blockIdx.x;
    const float* row = S + (size_t)i * NTOT;
    float lm = -1e30f;
    for (int j = threadIdx.x; j < NTOT; j += 256) lm = fmaxf(lm, row[j]);
    red[threadIdx.x] = lm; __syncthreads();
    for (int o = 128; o > 0; o >>= 1) { if (threadIdx.x < o) red[threadIdx.x] = fmaxf(red[threadIdx.x], red[threadIdx.x + o]); __syncthreads(); }
    const float m = red[0]; __syncthreads();
    float ls = 0.f;
    for (int j = threadIdx.x; j < NTOT; j += 256) ls += expf(row[j] - m);
    red[threadIdx.x] = ls; __syncthreads();
    for (int o = 128; o > 0; o >>= 1) { if (threadIdx.x < o) red[threadIdx.x] += red[threadIdx.x + o]; __syncthreads(); }
    const float inv = w / red[0];
    float* arow = adj + (size_t)i * NTOT;
    for (int j = threadIdx.x; j < NTOT; j += 256) arow[j] += expf(row[j] - m) * inv;
}

__global__ void k_mask(float* __restrict__ adj)
{
    int j = blockIdx.x * blockDim.x + threadIdx.x;   // 0..N_TG-1
    int i = blockIdx.y;                              // 0..N_TG-1
    if (j >= N_TG) return;
    if (i != j) adj[(size_t)(N_LM + i) * NTOT + (N_LM + j)] = 0.f;
}

// ---------------- per-row radix select of KSEL-th largest ----------------
__global__ __launch_bounds__(256)
void k_topk_thresh(const float* __restrict__ adj, float* __restrict__ thr)
{
    __shared__ unsigned int hist[256];
    __shared__ unsigned int prefix_s, rem_s;
    const int i = blockIdx.x;
    const float* row = adj + (size_t)i * NTOT;
    if (threadIdx.x == 0) { prefix_s = 0u; rem_s = KSEL; }
    __syncthreads();
    for (int pass = 3; pass >= 0; --pass) {
        hist[threadIdx.x] = 0u;
        __syncthreads();
        unsigned int pfx = prefix_s;
        for (int j = threadIdx.x; j < NTOT; j += 256) {
            unsigned int key = __float_as_uint(row[j]);     // all values >= 0
            unsigned int hb = (pass == 3) ? 0u : (key >> (8 * (pass + 1)));
            if (hb == pfx) atomicAdd(&hist[(key >> (8 * pass)) & 255u], 1u);
        }
        __syncthreads();
        if (threadIdx.x == 0) {
            unsigned int cum = 0; int chosen = 0;
            for (int b = 255; b >= 0; --b) {
                unsigned int h = hist[b];
                if (cum + h >= rem_s) { chosen = b; rem_s -= cum; break; }
                cum += h;
            }
            prefix_s = (prefix_s << 8) | (unsigned)chosen;
        }
        __syncthreads();
    }
    if (threadIdx.x == 0) thr[i] = __uint_as_float(prefix_s);
}

// ---------------- zero-preserving row softmax over selected entries ----------------
__global__ __launch_bounds__(256)
void k_row_norm(float* __restrict__ adj, unsigned short* __restrict__ adjbf,
                const float* __restrict__ thr)
{
    __shared__ float red[256];
    const int i = blockIdx.x;
    const float t = thr[i];
    float* row = adj + (size_t)i * NTOT;
    float lm = -1e30f;
    for (int j = threadIdx.x; j < NTOT; j += 256) { float v = row[j]; if (v >= t) lm = fmaxf(lm, v); }
    red[threadIdx.x] = lm; __syncthreads();
    for (int o = 128; o > 0; o >>= 1) { if (threadIdx.x < o) red[threadIdx.x] = fmaxf(red[threadIdx.x], red[threadIdx.x + o]); __syncthreads(); }
    const float m = red[0]; __syncthreads();
    float ls = 0.f;
    for (int j = threadIdx.x; j < NTOT; j += 256) { float v = row[j]; if (v >= t) ls += expf(v - m); }
    red[threadIdx.x] = ls; __syncthreads();
    for (int o = 128; o > 0; o >>= 1) { if (threadIdx.x < o) red[threadIdx.x] += red[threadIdx.x + o]; __syncthreads(); }
    const float inv = 1.f / red[0];
    unsigned short* brow = adjbf + (size_t)i * NTOT;
    for (int j = threadIdx.x; j < NTOT; j += 256) {
        float v = row[j];
        float o = (v >= t) ? expf(v - m) * inv : 0.f;
        row[j] = o;
        brow[j] = f2bf(o);
    }
}

// ---------------- VGAE / CNF pieces ----------------
__global__ void k_z0(const float* __restrict__ zmu, const float* __restrict__ zsigpre,
                     const float* __restrict__ eps, float* __restrict__ z0,
                     float* __restrict__ z, long long n)
{
    long long i = blockIdx.x * (long long)blockDim.x + threadIdx.x;
    if (i >= n) return;
    float sig = softplusf(zsigpre[i]) + 1e-10f;
    float v = zmu[i] + sig * eps[i];
    z0[i] = v; z[i] = v;
}

__global__ void k_cvec(const float* __restrict__ W1, const float* __restrict__ W2,
                       float* __restrict__ c)
{
    int j = threadIdx.x;
    if (j >= DZ) return;
    float s = 0.f;
    for (int i = 0; i < DZ; ++i) s += W1[i * DZ + j] * W2[j * DZ + i];
    c[j] = s;
}

__global__ __launch_bounds__(64)
void k_cnf_update(float* __restrict__ z, const float* __restrict__ H,
                  const float* __restrict__ F, const float* __restrict__ c,
                  float* __restrict__ lp, float dt)
{
    __shared__ float red[64];
    const int n = blockIdx.x, j = threadIdx.x;
    float h = H[(size_t)n * DZ + j];
    z[(size_t)n * DZ + j] += dt * F[(size_t)n * DZ + j];
    red[j] = (1.f - h * h) * c[j];
    __syncthreads();
    for (int o = 32; o > 0; o >>= 1) { if (j < o) red[j] += red[j + o]; __syncthreads(); }
    if (j == 0) lp[n] -= dt * red[0];
}

__global__ void k_ab(const float* __restrict__ z, const float* __restrict__ recW,
                     float* __restrict__ av, float* __restrict__ bv)
{
    int n = blockIdx.x * blockDim.x + threadIdx.x;
    if (n >= NTOT) return;
    float a = 0.f, b = 0.f;
    const float* zr = z + (size_t)n * DZ;
    for (int j = 0; j < DZ; ++j) { a += zr[j] * recW[j]; b += zr[j] * recW[DZ + j]; }
    av[n] = a; bv[n] = b;
}

// ---------------- reductions ----------------
__global__ __launch_bounds__(256)
void k_reduce(const float* __restrict__ x, long long n, float* __restrict__ acc, int sq)
{
    __shared__ float red[256];
    float s = 0.f;
    for (long long i = blockIdx.x * 256ll + threadIdx.x; i < n; i += (long long)gridDim.x * 256)
    { float v = x[i]; s += sq ? v * v : v; }
    red[threadIdx.x] = s; __syncthreads();
    for (int o = 128; o > 0; o >>= 1) { if (threadIdx.x < o) red[threadIdx.x] += red[threadIdx.x + o]; __syncthreads(); }
    if (threadIdx.x == 0) atomicAdd(acc, red[0]);
}

__global__ __launch_bounds__(256)
void k_sqdiff(const float* __restrict__ a, const float* __restrict__ b,
              long long n, float* __restrict__ acc)
{
    __shared__ float red[256];
    float s = 0.f;
    for (long long i = blockIdx.x * 256ll + threadIdx.x; i < n; i += (long long)gridDim.x * 256)
    { float d = a[i] - b[i]; s += d * d; }
    red[threadIdx.x] = s; __syncthreads();
    for (int o = 128; o > 0; o >>= 1) { if (threadIdx.x < o) red[threadIdx.x] += red[threadIdx.x + o]; __syncthreads(); }
    if (threadIdx.x == 0) atomicAdd(acc, red[0]);
}

__global__ __launch_bounds__(256)
void k_rec_loss(const float* __restrict__ adj, const float* __restrict__ av,
                const float* __restrict__ bv, const float* __restrict__ recb,
                float* __restrict__ acc)
{
    __shared__ float red[256];
    const float rb = recb[0];
    float s = 0.f;
    for (long long idx = blockIdx.x * 256ll + threadIdx.x; idx < NN; idx += (long long)gridDim.x * 256) {
        if (adj[idx] != 0.f) {
            int i = (int)(idx >> 12), j = (int)(idx & 4095);
            s += softplusf(av[j] + bv[i] + rb);
        }
    }
    red[threadIdx.x] = s; __syncthreads();
    for (int o = 128; o > 0; o >>= 1) { if (threadIdx.x < o) red[threadIdx.x] += red[threadIdx.x + o]; __syncthreads(); }
    if (threadIdx.x == 0) atomicAdd(acc, red[0]);
}

__global__ __launch_bounds__(256)
void k_colsum2(const float* __restrict__ Y, int rows, float* __restrict__ acc2)
{
    __shared__ float r0[256], r1[256];
    float s0 = 0.f, s1 = 0.f;
    for (int i = blockIdx.x * 256 + threadIdx.x; i < rows; i += gridDim.x * 256)
    { s0 += Y[(size_t)i * 2]; s1 += Y[(size_t)i * 2 + 1]; }
    r0[threadIdx.x] = s0; r1[threadIdx.x] = s1; __syncthreads();
    for (int o = 128; o > 0; o >>= 1) { if (threadIdx.x < o) { r0[threadIdx.x] += r0[threadIdx.x + o]; r1[threadIdx.x] += r1[threadIdx.x + o]; } __syncthreads(); }
    if (threadIdx.x == 0) { atomicAdd(&acc2[0], r0[0]); atomicAdd(&acc2[1], r1[0]); }
}

__global__ void k_build_hz(const float* __restrict__ hu, const float* __restrict__ z,
                           float* __restrict__ hz)
{
    long long idx = blockIdx.x * (long long)blockDim.x + threadIdx.x;
    long long tot = (long long)N_TG * (2 * DZ);
    if (idx >= tot) return;
    int r = (int)(idx / (2 * DZ)), c = (int)(idx % (2 * DZ));
    int gr = N_LM + r;
    hz[idx] = (c < DZ) ? hu[(size_t)gr * DZ + c] : z[(size_t)gr * DZ + (c - DZ)];
}

__global__ void k_finalize(float* __restrict__ out, const float* __restrict__ acc)
{
    int t = blockIdx.x * blockDim.x + threadIdx.x;
    if (t < N_TG * 2) {
        out[t] += acc[5 + (t & 1)] / (float)N_LM;
    } else if (t == N_TG * 2) {
        float g_loss = acc[0] / (float)((long long)NTOT * DIN);
        float rec    = acc[1] / (float)NN;
        float mzt    = acc[2] / (float)((long long)NTOT * DZ);
        float mz0    = acc[3] / (float)((long long)NTOT * DZ);
        float mlp    = acc[4] / (float)NTOT;
        float elbo   = rec - 0.5f * mzt + 0.5f * mz0 + mlp;
        out[t] = g_loss * 0.8f - elbo * 0.4f;
    }
}

// =====================================================================
// host-side orchestration
// =====================================================================
static inline size_t wsalloc(size_t& cur, size_t bytes) {
    size_t r = cur; cur += (bytes + 255) & ~(size_t)255; return r;
}

extern "C" void kernel_launch(void* const* d_in, const int* in_sizes, int n_in,
                              void* d_out, int out_size, void* d_ws, size_t ws_size,
                              hipStream_t stream)
{
    (void)in_sizes; (void)n_in; (void)out_size; (void)ws_size;
    // ---- inputs (setup_inputs dict order; params in make_params dict order)
    const float* lm_X  = (const float*)d_in[0];
    const float* lm_Y  = (const float*)d_in[1];
    const float* tg_X  = (const float*)d_in[2];
    const float* tg_Y  = (const float*)d_in[3];
    const float* lm_dl = (const float*)d_in[4];
    const float* tg_dl = (const float*)d_in[5];
    const float* eps   = (const float*)d_in[6];
    const float* dis_co_W = (const float*)d_in[7];
    const float* dis_co_b = (const float*)d_in[8];
    const float* kappa_W  = (const float*)d_in[9];
    const float* kappa_b  = (const float*)d_in[10];
    const float* Wq       = (const float*)d_in[11];
    const float* Wk       = (const float*)d_in[12];
    const float* enc1_W   = (const float*)d_in[13];
    const float* enc1_b   = (const float*)d_in[14];
    const float* enc2_W   = (const float*)d_in[15];
    const float* enc2_b   = (const float*)d_in[16];
    const float* dec_W    = (const float*)d_in[17];
    const float* dec_b    = (const float*)d_in[18];
    const float* gnn1_W   = (const float*)d_in[19];
    const float* gnn1_b   = (const float*)d_in[20];
    const float* gnn2_W   = (const float*)d_in[21];
    const float* gnn2_b   = (const float*)d_in[22];
    const float* mu_W     = (const float*)d_in[23];
    const float* mu_b     = (const float*)d_in[24];
    const float* var_W    = (const float*)d_in[25];
    const float* var_b    = (const float*)d_in[26];
    const float* rec_W    = (const float*)d_in[27];
    const float* rec_b    = (const float*)d_in[28];
    const float* cnf_W1   = (const float*)d_in[29];
    const float* cnf_b1   = (const float*)d_in[30];
    const float* cnf_W2   = (const float*)d_in[31];
    const float* cnf_b2   = (const float*)d_in[32];
    const float* pred1_W  = (const float*)d_in[33];
    const float* pred1_b  = (const float*)d_in[34];
    const float* pred2_W  = (const float*)d_in[35];
    const float* pred2_b  = (const float*)d_in[36];
    float* out = (float*)d_out;

    // ---- workspace carve-out
    char* ws = (char*)d_ws; size_t cur = 0;
    float*          adj    = (float*)(ws + wsalloc(cur, NN * 4));
    float*          Sbuf   = (float*)(ws + wsalloc(cur, NN * 4));
    unsigned short* adjbf  = (unsigned short*)(ws + wsalloc(cur, NN * 2));
    float*          Qf     = (float*)(ws + wsalloc(cur, (size_t)NTOT * QKDIM * 4));
    float*          Kf     = (float*)(ws + wsalloc(cur, (size_t)NTOT * QKDIM * 4));
    unsigned short* qbf    = (unsigned short*)(ws + wsalloc(cur, (size_t)NTOT * QKDIM * 2));
    unsigned short* kbf    = (unsigned short*)(ws + wsalloc(cur, (size_t)NTOT * QKDIM * 2));
    float*          Xc     = (float*)(ws + wsalloc(cur, (size_t)NTOT * DIN * 4));
    float*          Yc     = (float*)(ws + wsalloc(cur, (size_t)NTOT * 2 * 4));
    float*          dvec   = (float*)(ws + wsalloc(cur, NTOT * 4));
    float*          dist   = (float*)(ws + wsalloc(cur, NTOT * 4));
    float*          kap    = (float*)(ws + wsalloc(cur, NTOT * 4));
    float*          thr    = (float*)(ws + wsalloc(cur, NTOT * 4));
    unsigned short* XTbf   = (unsigned short*)(ws + wsalloc(cur, (size_t)DIN * NTOT * 2));
    float*          AX     = (float*)(ws + wsalloc(cur, (size_t)NTOT * DIN * 4));
    float*          h1     = (float*)(ws + wsalloc(cur, (size_t)NTOT * DAS * 4));
    unsigned short* h1Tbf  = (unsigned short*)(ws + wsalloc(cur, (size_t)DAS * NTOT * 2));
    float*          AH1    = (float*)(ws + wsalloc(cur, (size_t)NTOT * DAS * 4));
    float*          henc   = (float*)(ws + wsalloc(cur, (size_t)NTOT * DAS * 4));
    float*          xdec   = (float*)(ws + wsalloc(cur, (size_t)NTOT * DIN * 4));
    unsigned short* inpTbf = (unsigned short*)(ws + wsalloc(cur, (size_t)NPAD * NTOT * 2));
    float*          Ainp   = (float*)(ws + wsalloc(cur, (size_t)NTOT * NPAD * 4));
    float*          g1     = (float*)(ws + wsalloc(cur, (size_t)NTOT * DZ * 4));
    unsigned short* g1Tbf  = (unsigned short*)(ws + wsalloc(cur, (size_t)DZ * NTOT * 2));
    float*          Ag1    = (float*)(ws + wsalloc(cur, (size_t)NTOT * DZ * 4));
    float*          hu     = (float*)(ws + wsalloc(cur, (size_t)NTOT * DZ * 4));
    float*          zmu    = (float*)(ws + wsalloc(cur, (size_t)NTOT * DZ * 4));
    float*          z0b    = (float*)(ws + wsalloc(cur, (size_t)NTOT * DZ * 4));
    float*          zb     = (float*)(ws + wsalloc(cur, (size_t)NTOT * DZ * 4));
    float*          Hb     = (float*)(ws + wsalloc(cur, (size_t)NTOT * DZ * 4));
    float*          Fb     = (float*)(ws + wsalloc(cur, (size_t)NTOT * DZ * 4));
    float*          lp     = (float*)(ws + wsalloc(cur, NTOT * 4));
    float*          cvec   = (float*)(ws + wsalloc(cur, DZ * 4));
    float*          av     = (float*)(ws + wsalloc(cur, NTOT * 4));
    float*          bv     = (float*)(ws + wsalloc(cur, NTOT * 4));
    float*          hz     = (float*)(ws + wsalloc(cur, (size_t)N_TG * 2 * DZ * 4));
    float*          Pb     = (float*)(ws + wsalloc(cur, (size_t)N_TG * DINNER * 4));
    float*          acc    = (float*)(ws + wsalloc(cur, 8 * 4));

    const dim3 b256(256), b16(16, 16), bGemm(128);
    auto g1d = [](long long n) { return dim3((unsigned)((n + 255) / 256)); };

    // ---- init
    k_zero<<<g1d(8), b256, 0, stream>>>(acc, 8);
    k_zero<<<g1d(NTOT), b256, 0, stream>>>(lp, NTOT);

    // ---- concatenations
    k_concat_rows<<<g1d((long long)NTOT * DIN), b256, 0, stream>>>(lm_X, N_LM, tg_X, N_TG, DIN, Xc, 0);
    k_concat_rows<<<g1d((long long)NTOT * 2), b256, 0, stream>>>(lm_Y, N_LM, tg_Y, N_TG, 2, Yc, 1);
    k_concat_rows<<<g1d(NTOT), b256, 0, stream>>>(lm_dl, N_LM, tg_dl, N_TG, 1, dvec, 0);

    // ---- distance / kappa -> edge_nt into adj
    k_node_scalars<<<g1d(NTOT), b256, 0, stream>>>(Xc, dis_co_W, dis_co_b, kappa_W, kappa_b, dvec, dist, kap);
    k_edge_nt<<<g1d(NN), b256, 0, stream>>>(dist, kap, adj);

    // ---- attention: q/k, bf16 convert (1/sqrt(DK) folded into q), per-head WMMA + softmax
    sgemm_bias_act<<<dim3(QKDIM / 16, NTOT / 16), b16, 0, stream>>>(Xc, DIN, Wq, nullptr, Qf, QKDIM, NTOT, QKDIM, DIN, 0);
    sgemm_bias_act<<<dim3(QKDIM / 16, NTOT / 16), b16, 0, stream>>>(Xc, DIN, Wk, nullptr, Kf, QKDIM, NTOT, QKDIM, DIN, 0);
    k_f32_to_bf16<<<g1d((long long)NTOT * QKDIM), b256, 0, stream>>>(Qf, qbf, (long long)NTOT * QKDIM, 0.08838834764831845f);
    k_f32_to_bf16<<<g1d((long long)NTOT * QKDIM), b256, 0, stream>>>(Kf, kbf, (long long)NTOT * QKDIM, 1.0f);
    for (int h = 0; h < NHEAD; ++h) {
        wmma_gemm_bf16_tdm<<<dim3(NTOT / 64, NTOT / 64), bGemm, 0, stream>>>(
            qbf + h * DKH, QKDIM, kbf + h * DKH, QKDIM, Sbuf, NTOT, DKH, 1.0f);
        k_softmax_acc<<<dim3(NTOT), b256, 0, stream>>>(Sbuf, adj, 0.25f);
    }

    // ---- mask, top-k threshold, zero-preserving row softmax (+ bf16 copy)
    k_mask<<<dim3((N_TG + 255) / 256, N_TG), b256, 0, stream>>>(adj);
    k_topk_thresh<<<dim3(NTOT), b256, 0, stream>>>(adj, thr);
    k_row_norm<<<dim3(NTOT), b256, 0, stream>>>(adj, adjbf, thr);

    // ---- autoencoder branch
    k_transpose_bf16<<<g1d((long long)NTOT * DIN), b256, 0, stream>>>(Xc, XTbf, NTOT, DIN);
    wmma_gemm_bf16_tdm<<<dim3(DIN / 64, NTOT / 64), bGemm, 0, stream>>>(adjbf, NTOT, XTbf, NTOT, AX, DIN, NTOT, 1.0f);
    sgemm_bias_act<<<dim3(DAS / 16, NTOT / 16), b16, 0, stream>>>(AX, DIN, enc1_W, enc1_b, h1, DAS, NTOT, DAS, DIN, 1);
    k_transpose_bf16<<<g1d((long long)NTOT * DAS), b256, 0, stream>>>(h1, h1Tbf, NTOT, DAS);
    wmma_gemm_bf16_tdm<<<dim3(DAS / 64, NTOT / 64), bGemm, 0, stream>>>(adjbf, NTOT, h1Tbf, NTOT, AH1, DAS, NTOT, 1.0f);
    sgemm_bias_act<<<dim3(DAS / 16, NTOT / 16), b16, 0, stream>>>(AH1, DAS, enc2_W, enc2_b, henc, DAS, NTOT, DAS, DAS, 0);
    sgemm_bias_act<<<dim3(DIN / 16, NTOT / 16), b16, 0, stream>>>(henc, DAS, dec_W, dec_b, xdec, DIN, NTOT, DIN, DAS, 0);
    k_sqdiff<<<dim3(512), b256, 0, stream>>>(xdec, Xc, (long long)NTOT * DIN, acc + 0);

    // ---- GNN branch (inp = [X, Y], padded 66 -> NPAD for WMMA)
    k_build_inpT<<<g1d((long long)NPAD * NTOT), b256, 0, stream>>>(Xc, Yc, inpTbf);
    wmma_gemm_bf16_tdm<<<dim3(NPAD / 64, NTOT / 64), bGemm, 0, stream>>>(adjbf, NTOT, inpTbf, NTOT, Ainp, NPAD, NTOT, 1.0f);
    sgemm_bias_act<<<dim3(DZ / 16, NTOT / 16), b16, 0, stream>>>(Ainp, NPAD, gnn1_W, gnn1_b, g1, DZ, NTOT, DZ, DIN + 2, 1);
    k_transpose_bf16<<<g1d((long long)NTOT * DZ), b256, 0, stream>>>(g1, g1Tbf, NTOT, DZ);
    wmma_gemm_bf16_tdm<<<dim3(DZ / 64, NTOT / 64), bGemm, 0, stream>>>(adjbf, NTOT, g1Tbf, NTOT, Ag1, DZ, NTOT, 1.0f);
    sgemm_bias_act<<<dim3(DZ / 16, NTOT / 16), b16, 0, stream>>>(Ag1, DZ, gnn2_W, gnn2_b, hu, DZ, NTOT, DZ, DZ, 0);

    // ---- VGAE reparameterization
    sgemm_bias_act<<<dim3(DZ / 16, NTOT / 16), b16, 0, stream>>>(hu, DZ, mu_W, mu_b, zmu, DZ, NTOT, DZ, DZ, 0);
    sgemm_bias_act<<<dim3(DZ / 16, NTOT / 16), b16, 0, stream>>>(hu, DZ, var_W, var_b, Fb, DZ, NTOT, DZ, DZ, 0);
    k_z0<<<g1d((long long)NTOT * DZ), b256, 0, stream>>>(zmu, Fb, eps, z0b, zb, (long long)NTOT * DZ);

    // ---- CNF (8 steps)
    k_cvec<<<dim3(1), dim3(64), 0, stream>>>(cnf_W1, cnf_W2, cvec);
    const float dt = 1.0f / CNF_STEPS;
    for (int s = 0; s < CNF_STEPS; ++s) {
        sgemm_bias_act<<<dim3(DZ / 16, NTOT / 16), b16, 0, stream>>>(zb, DZ, cnf_W1, cnf_b1, Hb, DZ, NTOT, DZ, DZ, 2);
        sgemm_bias_act<<<dim3(DZ / 16, NTOT / 16), b16, 0, stream>>>(Hb, DZ, cnf_W2, cnf_b2, Fb, DZ, NTOT, DZ, DZ, 0);
        k_cnf_update<<<dim3(NTOT), dim3(64), 0, stream>>>(zb, Hb, Fb, cvec, lp, dt);
    }

    // ---- reconstruction loss + ELBO pieces
    k_ab<<<g1d(NTOT), b256, 0, stream>>>(zb, rec_W, av, bv);
    k_rec_loss<<<dim3(1024), b256, 0, stream>>>(adj, av, bv, rec_b, acc + 1);
    k_reduce<<<dim3(512), b256, 0, stream>>>(zb, (long long)NTOT * DZ, acc + 2, 1);
    k_reduce<<<dim3(512), b256, 0, stream>>>(z0b, (long long)NTOT * DZ, acc + 3, 1);
    k_reduce<<<dim3(64), b256, 0, stream>>>(lp, NTOT, acc + 4, 0);
    k_colsum2<<<dim3(16), b256, 0, stream>>>(lm_Y, N_LM, acc + 5);

    // ---- prediction head (only target rows) + final loss
    k_build_hz<<<g1d((long long)N_TG * 2 * DZ), b256, 0, stream>>>(hu, zb, hz);
    sgemm_bias_act<<<dim3(DINNER / 16, N_TG / 16), b16, 0, stream>>>(hz, 2 * DZ, pred1_W, pred1_b, Pb, DINNER, N_TG, DINNER, 2 * DZ, 1);
    sgemm_bias_act<<<dim3(1, N_TG / 16), b16, 0, stream>>>(Pb, DINNER, pred2_W, pred2_b, out, DOUT, N_TG, DOUT, DINNER, 0);
    k_finalize<<<g1d(N_TG * 2 + 1), b256, 0, stream>>>(out, acc);
}
